// PocEnc_781684048045
// MI455X (gfx1250) — matrix operand order repaired and checked by reading, not compiled, not run on hardware
//
#include <hip/hip_runtime.h>
#include <math.h>

typedef __attribute__((ext_vector_type(16))) __bf16 v16bf;
typedef __attribute__((ext_vector_type(8)))  float  v8f;

#define NGRAPH 8
#define NEDGE  65536
#define NODES0 8192
#define HIDDIM 64
#define OUTDIM 512

// ---------------------------------------------------------------------------
// elementwise helpers
// ---------------------------------------------------------------------------
__global__ void k_fill_f32(float* p, float v, long n) {
  long i = blockIdx.x * (long)blockDim.x + threadIdx.x;
  if (i < n) p[i] = v;
}
__global__ void k_fill_i32(int* p, int v, long n) {
  long i = blockIdx.x * (long)blockDim.x + threadIdx.x;
  if (i < n) p[i] = v;
}
__global__ void k_init_rows(float* h, const float* __restrict__ bias, long n, int d) {
  long i = blockIdx.x * (long)blockDim.x + threadIdx.x;
  if (i < n) h[i] = bias[(int)(i % d)];
}
__global__ void k_leaky(float* h, long n, float slope) {
  long i = blockIdx.x * (long)blockDim.x + threadIdx.x;
  if (i < n) { float v = h[i]; h[i] = v > 0.f ? v : slope * v; }
}

// ---------------------------------------------------------------------------
// WMMA tile packing (bf16)
// A operand (activations, M x K): lane = row(M)%16, lane>=16 selects second K
// half; element j -> k = (j<8 ? j : j+8) + 8*laneHi   (ISA 7.12.2, 16-bit A)
// B operand (weights, K x N): lane = col(N)%16; element j -> k = (j<8 ? j : j+8) + 8*laneHi
// Out-of-range rows/cols are zero-filled, so padded tiles are benign.
// ---------------------------------------------------------------------------
__global__ void k_pack_a(const float* __restrict__ X, __bf16* __restrict__ P,
                         int M, int K, int Kt, long total) {
  long idx = blockIdx.x * (long)blockDim.x + threadIdx.x;
  if (idx >= total) return;
  int  j    = (int)(idx & 15);
  int  lane = (int)((idx >> 4) & 31);
  long tile = idx >> 9;
  int  kt   = (int)(tile % Kt);
  int  mt   = (int)(tile / Kt);
  int  laneHi = lane >> 4;
  int  m = mt * 16 + (lane & 15);
  int  k = kt * 32 + ((j < 8 ? j : j + 8) + (laneHi ? 8 : 0));
  float v = (m < M && k < K) ? X[(long)m * K + k] : 0.f;
  P[idx] = (__bf16)v;
}
__global__ void k_pack_b(const float* __restrict__ W, __bf16* __restrict__ P,
                         int K, int N, int Kt, long total) {
  long idx = blockIdx.x * (long)blockDim.x + threadIdx.x;
  if (idx >= total) return;
  int  j    = (int)(idx & 15);
  int  lane = (int)((idx >> 4) & 31);
  long tile = idx >> 9;
  int  kt   = (int)(tile % Kt);
  int  nt   = (int)(tile / Kt);
  int  laneHi = lane >> 4;
  int  n = nt * 16 + (lane & 15);
  int  k = kt * 32 + ((j < 8 ? j : j + 8) + (laneHi ? 8 : 0));
  float v = (k < K && n < N) ? W[(long)k * N + n] : 0.f;
  P[idx] = (__bf16)v;
}

// ---------------------------------------------------------------------------
// Tiled WMMA GEMM: C[M,N] = act(A@B + bias). One wave per (mt, 4 n-tiles).
// NtPad is a multiple of 4 (B buffer padded with zero tiles), so the inner
// K-loop is branch-free: 8 clause-able b128 loads + 4 back-to-back WMMAs.
// act: 0=none, 1=SiLU
// ---------------------------------------------------------------------------
__global__ void k_gemm(const __bf16* __restrict__ PA, const __bf16* __restrict__ PB,
                       float* __restrict__ C, const float* __restrict__ bias,
                       int M, int Nact, int Kt, int Mt, int NtPad, int act,
                       int rowOffset) {
  int wid  = blockIdx.x * (blockDim.x >> 5) + (threadIdx.x >> 5);
  int lane = threadIdx.x & 31;
  int NtG  = NtPad >> 2;
  if (wid >= Mt * NtG) return;
  int mt = wid / NtG, ntg = wid % NtG;
  int nt0 = ntg * 4;
  v8f acc0 = {}, acc1 = {}, acc2 = {}, acc3 = {};
  const __bf16* pa = PA + ((size_t)mt * Kt << 9) + lane * 16;
  for (int kt = 0; kt < Kt; ++kt) {
    v16bf a = *(const v16bf*)(pa + ((size_t)kt << 9));
    if (kt + 1 < Kt)
      __builtin_prefetch(pa + ((size_t)(kt + 1) << 9), 0, 3);
    v16bf b0 = *(const v16bf*)(PB + ((size_t)((nt0 + 0) * Kt + kt) << 9) + lane * 16);
    v16bf b1 = *(const v16bf*)(PB + ((size_t)((nt0 + 1) * Kt + kt) << 9) + lane * 16);
    v16bf b2 = *(const v16bf*)(PB + ((size_t)((nt0 + 2) * Kt + kt) << 9) + lane * 16);
    v16bf b3 = *(const v16bf*)(PB + ((size_t)((nt0 + 3) * Kt + kt) << 9) + lane * 16);
    acc0 = __builtin_amdgcn_wmma_f32_16x16x32_bf16(false, a, false, b0, (short)0, acc0, false, false);
    acc1 = __builtin_amdgcn_wmma_f32_16x16x32_bf16(false, a, false, b1, (short)0, acc1, false, false);
    acc2 = __builtin_amdgcn_wmma_f32_16x16x32_bf16(false, a, false, b2, (short)0, acc2, false, false);
    acc3 = __builtin_amdgcn_wmma_f32_16x16x32_bf16(false, a, false, b3, (short)0, acc3, false, false);
  }
  int laneN = lane & 15, laneHi = lane >> 4;
#define EPI(ii, ACC)                                                            \
  {                                                                             \
    int n = (nt0 + ii) * 16 + laneN;                                            \
    if (n < Nact) {                                                             \
      float bv = bias ? bias[n] : 0.f;                                          \
      _Pragma("unroll") for (int r = 0; r < 8; ++r) {                           \
        int m = mt * 16 + r + laneHi * 8;                                       \
        if (m < M) {                                                            \
          float v = ACC[r] + bv;                                                \
          if (act == 1) v = v / (1.f + __expf(-v));                             \
          C[(size_t)(rowOffset + m) * Nact + n] = v;                            \
        }                                                                       \
      }                                                                         \
    }                                                                           \
  }
  EPI(0, acc0) EPI(1, acc1) EPI(2, acc2) EPI(3, acc3)
#undef EPI
}

// ---------------------------------------------------------------------------
// Fused GATv2 edge logits: per 16-edge tile, xe = ea@We via WMMA (K=64),
// add gathered xl[src]+xr[dst], leaky_relu(0.2), dot with att, reduce per
// (edge, head) via LDS ds_add_f32. Masked edges -> -inf.
// ---------------------------------------------------------------------------
__global__ void k_edge_logits(const __bf16* __restrict__ PEA,
                              const __bf16* __restrict__ PWe,
                              const float* __restrict__ xl, const float* __restrict__ xr,
                              const int* __restrict__ src, const int* __restrict__ dst,
                              const int* __restrict__ msk, const float* __restrict__ att,
                              float* __restrict__ logits, int dout, int Nt) {
  __shared__ float red[8 * 64];
  int wv = threadIdx.x >> 5, lane = threadIdx.x & 31;
  int t = blockIdx.x * 8 + wv;                 // edge tile (E/16 = 4096, exact)
  float* rw = &red[wv * 64];
  rw[lane] = 0.f; rw[lane + 32] = 0.f;
  __syncthreads();

  int laneN = lane & 15, laneHi = lane >> 4;
  v16bf a0 = *(const v16bf*)(PEA + ((size_t)(t * 2 + 0) << 9) + lane * 16);
  v16bf a1 = *(const v16bf*)(PEA + ((size_t)(t * 2 + 1) << 9) + lane * 16);
  int N4 = 4 * dout;
  int s8[8], d8[8];
#pragma unroll
  for (int r = 0; r < 8; ++r) {
    int e = t * 16 + r + laneHi * 8;
    s8[r] = src[e]; d8[r] = dst[e];
  }
  for (int nt = 0; nt < Nt; ++nt) {
    v16bf b0 = *(const v16bf*)(PWe + ((size_t)(nt * 2 + 0) << 9) + lane * 16);
    v16bf b1 = *(const v16bf*)(PWe + ((size_t)(nt * 2 + 1) << 9) + lane * 16);
    v8f acc = {};
    acc = __builtin_amdgcn_wmma_f32_16x16x32_bf16(false, a0, false, b0, (short)0, acc, false, false);
    acc = __builtin_amdgcn_wmma_f32_16x16x32_bf16(false, a1, false, b1, (short)0, acc, false, false);
    int col = nt * 16 + laneN;
    if (col < N4) {
      int hh = col / dout;        // att is [HEADS, dout] row-major -> flat idx
      float av = att[col];
#pragma unroll
      for (int r = 0; r < 8; ++r) {
        float v = acc[r] + xl[(size_t)s8[r] * N4 + col] + xr[(size_t)d8[r] * N4 + col];
        v = v > 0.f ? v : 0.2f * v;
        atomicAdd(&rw[(r + laneHi * 8) * 4 + hh], v * av);
      }
    }
  }
  __syncthreads();
#pragma unroll
  for (int q = 0; q < 2; ++q) {
    int o = lane * 2 + q;         // 64 outputs: (edge-in-tile, head)
    int m = o >> 2, h = o & 3;
    int e = t * 16 + m;
    logits[e * 4 + h] = msk[e] ? rw[o] : -__builtin_inff();
  }
}

// ---------------------------------------------------------------------------
// softmax over in-edges (segment max / exp / den), then aggregation
// ---------------------------------------------------------------------------
__global__ void k_segmax(const float* __restrict__ logits, const int* __restrict__ dst,
                         float* mmax, int E) {
  int i = blockIdx.x * blockDim.x + threadIdx.x;
  if (i >= E * 4) return;
  int e = i >> 2, h = i & 3;
  float v = logits[i];
  float* addr = &mmax[dst[e] * 4 + h];
  float old = *addr;
  while (v > old) {
    int assumed = __float_as_int(old);
    int prev = atomicCAS((int*)addr, assumed, __float_as_int(v));
    if (prev == assumed) break;
    old = __int_as_float(prev);
  }
}
__global__ void k_expden(const float* __restrict__ logits, const int* __restrict__ dst,
                         const float* __restrict__ mmax, float* __restrict__ ex,
                         float* den, int E) {
  int i = blockIdx.x * blockDim.x + threadIdx.x;
  if (i >= E * 4) return;
  int e = i >> 2, h = i & 3;
  float md = mmax[dst[e] * 4 + h];
  if (md < -1e38f) md = 0.f;                 // isfinite(m) ? m : 0
  float v = __expf(logits[i] - md);          // -inf -> 0 (masked)
  ex[i] = v;
  atomicAdd(&den[dst[e] * 4 + h], v);
}
__global__ void k_aggregate(const float* __restrict__ xl, const float* __restrict__ ex,
                            const float* __restrict__ den, const int* __restrict__ src,
                            const int* __restrict__ dst, float* hnew, int dout, long total) {
  long idx = blockIdx.x * (long)blockDim.x + threadIdx.x;
  if (idx >= total) return;
  int e = (int)(idx / dout), d = (int)(idx % dout);
  int s = src[e], t = dst[e];
  int N4 = 4 * dout;
  float a = 0.f;
#pragma unroll
  for (int h = 0; h < 4; ++h) {
    float dn = fmaxf(den[t * 4 + h], 1e-16f);
    a += xl[(size_t)s * N4 + h * dout + d] * (ex[e * 4 + h] / dn);
  }
  atomicAdd(&hnew[(size_t)t * dout + d], 0.25f * a);   // head mean folded in
}

// ---------------------------------------------------------------------------
// global mean pool + gmlp (Linear->LayerNorm->SiLU, accumulate graph_out)
// ---------------------------------------------------------------------------
__global__ void k_meanpool(const float* __restrict__ h, float* pooled, int n_cur, int dout) {
  int idx = blockIdx.x * blockDim.x + threadIdx.x;
  if (idx >= NGRAPH * dout) return;
  int b = idx / dout, d = idx % dout;
  float s = 0.f;
  for (int n = 0; n < n_cur; ++n) s += h[((size_t)b * n_cur + n) * dout + d];
  pooled[idx] = s / (float)n_cur;
}
__global__ void k_gmlp(const float* __restrict__ pooled, const float* __restrict__ W,
                       const float* __restrict__ bs, const float* __restrict__ g,
                       const float* __restrict__ be, float* gout, int din) {
  __shared__ float red[512];
  int b = blockIdx.x, o = threadIdx.x;
  const float* x = pooled + (size_t)b * din;
  float acc = bs[o];
  for (int d = 0; d < din; ++d) acc += x[d] * W[(size_t)d * 512 + o];
  red[o] = acc; __syncthreads();
  for (int s = 256; s > 0; s >>= 1) { if (o < s) red[o] += red[o + s]; __syncthreads(); }
  float mu = red[0] / 512.f; __syncthreads();
  float dv = acc - mu;
  red[o] = dv * dv; __syncthreads();
  for (int s = 256; s > 0; s >>= 1) { if (o < s) red[o] += red[o + s]; __syncthreads(); }
  float var = red[0] / 512.f;
  float y = dv * rsqrtf(var + 1e-5f) * g[o] + be[o];
  y = y / (1.f + __expf(-y));
  gout[b * 512 + o] += y;
}

// ---------------------------------------------------------------------------
// SAGPool: GraphConv scorer + per-graph top-k (rank by count) + gather + remap
// ---------------------------------------------------------------------------
__global__ void k_aggpool(const float* __restrict__ h, const int* __restrict__ src,
                          const int* __restrict__ dst, const int* __restrict__ msk,
                          float* agg, int dout, long total) {
  long i = blockIdx.x * (long)blockDim.x + threadIdx.x;
  if (i >= total) return;
  int e = (int)(i / dout), d = (int)(i % dout);
  if (msk[e]) atomicAdd(&agg[(size_t)dst[e] * dout + d], h[(size_t)src[e] * dout + d]);
}
__global__ void k_score(const float* __restrict__ agg, const float* __restrict__ h,
                        const float* __restrict__ Wrel, const float* __restrict__ Wroot,
                        const float* __restrict__ brel, float* score, int Nn, int d) {
  int n = blockIdx.x * blockDim.x + threadIdx.x;
  if (n >= Nn) return;
  const float* a = agg + (size_t)n * d;
  const float* hh = h + (size_t)n * d;
  float s = brel[0];
  for (int i = 0; i < d; ++i) s += a[i] * Wrel[i] + hh[i] * Wroot[i];
  score[n] = s;
}
__global__ void k_topk(const float* __restrict__ score, int* inv, int n_cur, int k) {
  __shared__ float s[1024];
  int g = blockIdx.x;
  for (int i = threadIdx.x; i < n_cur; i += blockDim.x) s[i] = score[g * n_cur + i];
  __syncthreads();
  for (int i = threadIdx.x; i < n_cur; i += blockDim.x) {
    float si = s[i];
    int cnt = 0;
    for (int j = 0; j < n_cur; ++j) {
      float sj = s[j];
      cnt += (sj > si) || (sj == si && j < i);   // jax top_k tie-break
    }
    inv[g * n_cur + i] = (cnt < k) ? cnt : -1;
  }
}
__global__ void k_poolgather(const float* __restrict__ hin, const float* __restrict__ score,
                             const int* __restrict__ inv, float* hout,
                             int n_cur, int k, int d, long total) {
  long i = blockIdx.x * (long)blockDim.x + threadIdx.x;
  if (i >= total) return;
  long n = i / d; int c = (int)(i % d);
  int r = inv[n];
  if (r < 0) return;
  int g = (int)(n / n_cur);
  hout[((size_t)(g * k + r)) * d + c] = hin[i] * tanhf(score[n]);
}
__global__ void k_remap(const int* __restrict__ so, const int* __restrict__ dsto,
                        const int* __restrict__ msko, const int* __restrict__ inv,
                        int* sn, int* dn, int* mn, int n_cur, int k, int E) {
  int e = blockIdx.x * blockDim.x + threadIdx.x;
  if (e >= E) return;
  int s = so[e], d = dsto[e];
  int ns = inv[s], nd = inv[d];
  int gs = s / n_cur, gd = d / n_cur;
  int val = msko[e] && ns >= 0 && nd >= 0;
  sn[e] = val ? gs * k + ns : 0;
  dn[e] = val ? gd * k + nd : 0;
  mn[e] = val;
}

__global__ void k_fc(const float* __restrict__ gin, const float* __restrict__ W,
                     const float* __restrict__ bs, float* out) {
  int b = blockIdx.x, o = threadIdx.x;
  float a = bs[o];
  for (int d = 0; d < 512; ++d) a += gin[b * 512 + d] * W[(size_t)d * 512 + o];
  a = a / (1.f + __expf(-a));
  out[b * 512 + o] = a;
}

// ---------------------------------------------------------------------------
// host orchestration
// ---------------------------------------------------------------------------
static inline char* carve(char*& p, size_t bytes) {
  char* r = p; p += (bytes + 255) & ~(size_t)255; return r;
}
static inline unsigned cdiv(long a, long b) { return (unsigned)((a + b - 1) / b); }

extern "C" void kernel_launch(void* const* d_in, const int* in_sizes, int n_in,
                              void* d_out, int out_size, void* d_ws, size_t ws_size,
                              hipStream_t stream) {
  // Input map: top-level insertion order, params via jax-pytree sorted keys:
  //   edge_dense{W,b}, fc{W,b}, gat[i]{We,Wl,Wr,att,bias,bl,br},
  //   gmlp[i]{W,b,be,g}, node_dense{W,b}, pool[i]{Wrel,Wroot,brel}
  const float* x     = (const float*)d_in[0];
  const int*   ei    = (const int*)d_in[1];
  const float* eattr = (const float*)d_in[2];
  const float* edW = (const float*)d_in[4];  const float* edb = (const float*)d_in[5];
  const float* fcW = (const float*)d_in[6];  const float* fcb = (const float*)d_in[7];
  struct GatP { const float *We, *Wl, *Wr, *att, *bias, *bl, *br; } gat[3];
  for (int i = 0; i < 3; ++i) {
    int o = 8 + i * 7;
    gat[i] = { (const float*)d_in[o], (const float*)d_in[o+1], (const float*)d_in[o+2],
               (const float*)d_in[o+3], (const float*)d_in[o+4], (const float*)d_in[o+5],
               (const float*)d_in[o+6] };
  }
  struct GmlpP { const float *W, *b, *be, *g; } gm[3];
  for (int i = 0; i < 3; ++i) {
    int o = 29 + i * 4;
    gm[i] = { (const float*)d_in[o], (const float*)d_in[o+1], (const float*)d_in[o+2],
              (const float*)d_in[o+3] };
  }
  const float* ndW = (const float*)d_in[41]; const float* ndb = (const float*)d_in[42];
  struct PoolP { const float *Wrel, *Wroot, *brel; } pl[2] = {
    { (const float*)d_in[43], (const float*)d_in[44], (const float*)d_in[45] },
    { (const float*)d_in[46], (const float*)d_in[47], (const float*)d_in[48] } };

  const int* src0 = ei;
  const int* dst0 = ei + NEDGE;

  // workspace carving
  char* wp = (char*)d_ws;
  float*  H_A   = (float*) carve(wp, (size_t)NODES0 * 512 * 4);
  float*  H_B   = (float*) carve(wp, (size_t)NODES0 * 512 * 4);
  float*  EA    = (float*) carve(wp, (size_t)NEDGE * 64 * 4);
  __bf16* PEA   = (__bf16*)carve(wp, (size_t)4096 * 2 * 512 * 2);
  __bf16* PA    = (__bf16*)carve(wp, (size_t)24 << 20);
  __bf16* PB_L  = (__bf16*)carve(wp, (size_t)2 << 20);
  __bf16* PB_R  = (__bf16*)carve(wp, (size_t)2 << 20);
  __bf16* PB_E  = (__bf16*)carve(wp, (size_t)512 << 10);
  float*  XL    = (float*) carve(wp, (size_t)28 << 20);
  float*  XR    = (float*) carve(wp, (size_t)28 << 20);
  float*  LOGIT = (float*) carve(wp, (size_t)NEDGE * 4 * 4);
  float*  EX    = (float*) carve(wp, (size_t)NEDGE * 4 * 4);
  float*  MMAX  = (float*) carve(wp, (size_t)NODES0 * 4 * 4);
  float*  DEN   = (float*) carve(wp, (size_t)NODES0 * 4 * 4);
  float*  AGG   = (float*) carve(wp, (size_t)8 << 20);
  float*  SCORE = (float*) carve(wp, (size_t)NODES0 * 4);
  int*    INV   = (int*)   carve(wp, (size_t)NODES0 * 4);
  int*    MSK_A = (int*)   carve(wp, (size_t)NEDGE * 4);
  int*    SRC_B = (int*)   carve(wp, (size_t)NEDGE * 4);
  int*    DST_B = (int*)   carve(wp, (size_t)NEDGE * 4);
  int*    MSK_B = (int*)   carve(wp, (size_t)NEDGE * 4);
  int*    SRC_C = (int*)   carve(wp, (size_t)NEDGE * 4);
  int*    DST_C = (int*)   carve(wp, (size_t)NEDGE * 4);
  int*    MSK_C = (int*)   carve(wp, (size_t)NEDGE * 4);
  float*  POOLED= (float*) carve(wp, (size_t)NGRAPH * 512 * 4);
  float*  GOUT  = (float*) carve(wp, (size_t)NGRAPH * 512 * 4);

  auto packA = [&](const float* X, __bf16* P, int M, int K) {
    int Kt = (K + 31) / 32, Mt = (M + 15) / 16;
    long total = (long)Mt * Kt * 512;
    k_pack_a<<<cdiv(total, 256), 256, 0, stream>>>(X, P, M, K, Kt, total);
  };
  // pack B with the tile count padded to a multiple of 4 (zero tiles) so the
  // GEMM K-loop needs no per-tile bounds checks.
  auto packB = [&](const float* W, __bf16* P, int K, int N) {
    int Kt = (K + 31) / 32, Nt = (N + 15) / 16;
    int NtPad = (Nt + 3) & ~3;
    long total = (long)NtPad * Kt * 512;
    k_pack_b<<<cdiv(total, 256), 256, 0, stream>>>(W, P, K, N, Kt, total);
  };
  auto gemm = [&](const __bf16* PAp, const __bf16* PBp, float* C, const float* bias,
                  int M, int N, int K, int act, int rowOff) {
    int Kt = (K + 31) / 32, Mt = (M + 15) / 16, Nt = (N + 15) / 16;
    int NtPad = (Nt + 3) & ~3;
    long waves = (long)Mt * (NtPad >> 2);
    k_gemm<<<cdiv(waves, 8), 256, 0, stream>>>(PAp, PBp, C, bias, M, N, Kt, Mt,
                                               NtPad, act, rowOff);
  };

  // ---- init ----
  k_fill_i32<<<cdiv(NEDGE, 256), 256, 0, stream>>>(MSK_A, 1, NEDGE);
  k_fill_f32<<<cdiv(NGRAPH * 512, 256), 256, 0, stream>>>(GOUT, 0.f, NGRAPH * 512);

  // ---- node dense: h0 = silu(x@W + b), [8192, 64] ----
  packA(x, PA, NODES0, 1324);
  packB(ndW, PB_L, 1324, 64);
  gemm(PA, PB_L, H_A, ndb, NODES0, 64, 1324, 1, 0);

  // ---- edge dense in 4 row chunks: ea = silu(edge_attr@W + b), [65536, 64] ----
  packB(edW, PB_L, 434, 64);
  for (int c = 0; c < 4; ++c) {
    packA(eattr + (size_t)c * 16384 * 434, PA, 16384, 434);
    gemm(PA, PB_L, EA, edb, 16384, 64, 434, 1, c * 16384);
  }
  packA(EA, PEA, NEDGE, 64);   // WMMA A-tiles for ea, reused by all 3 layers

  const int dims[4] = { 64, 213, 362, 512 };
  int n_cur = 1024;
  const int* csrc = src0; const int* cdst = dst0; const int* cmsk = MSK_A;
  float* hcur = H_A;

  for (int i = 0; i < 3; ++i) {
    int din = dims[i], dout = dims[i + 1];
    int Nn = NGRAPH * n_cur, N4 = 4 * dout;
    int Nt4 = (N4 + 15) / 16;

    // xl = h@Wl + bl ; xr = h@Wr + br  (WMMA)
    packA(hcur, PA, Nn, din);
    packB(gat[i].Wl, PB_L, din, N4);
    packB(gat[i].Wr, PB_R, din, N4);
    packB(gat[i].We, PB_E, 64, N4);
    gemm(PA, PB_L, XL, gat[i].bl, Nn, N4, din, 0, 0);
    gemm(PA, PB_R, XR, gat[i].br, Nn, N4, din, 0, 0);

    // fused edge logits (xe WMMA + gather + leaky + att reduce)
    k_fill_f32<<<cdiv((long)Nn * 4, 256), 256, 0, stream>>>(MMAX, -__builtin_inff(),
                                                            (long)Nn * 4);
    k_fill_f32<<<cdiv((long)Nn * 4, 256), 256, 0, stream>>>(DEN, 0.f, (long)Nn * 4);
    k_edge_logits<<<512, 256, 0, stream>>>(PEA, PB_E, XL, XR, csrc, cdst, cmsk,
                                           gat[i].att, LOGIT, dout, Nt4);
    k_segmax<<<cdiv(NEDGE * 4, 256), 256, 0, stream>>>(LOGIT, cdst, MMAX, NEDGE);
    k_expden<<<cdiv(NEDGE * 4, 256), 256, 0, stream>>>(LOGIT, cdst, MMAX, EX, DEN, NEDGE);

    // h_gat = segment_sum(xl[src]*alpha).mean(heads) + bias  -> H_B
    k_init_rows<<<cdiv((long)Nn * dout, 256), 256, 0, stream>>>(H_B, gat[i].bias,
                                                                (long)Nn * dout, dout);
    k_aggregate<<<cdiv((long)NEDGE * dout, 256), 256, 0, stream>>>(XL, EX, DEN, csrc,
                                                                   cdst, H_B, dout,
                                                                   (long)NEDGE * dout);
    // graph_out += gmlp(mean_pool(h_gat))
    k_meanpool<<<cdiv(NGRAPH * dout, 256), 256, 0, stream>>>(H_B, POOLED, n_cur, dout);
    k_gmlp<<<NGRAPH, 512, 0, stream>>>(POOLED, gm[i].W, gm[i].b, gm[i].g, gm[i].be,
                                       GOUT, dout);

    if (i < 2) {
      k_leaky<<<cdiv((long)Nn * dout, 256), 256, 0, stream>>>(H_B, (long)Nn * dout, 0.01f);
      k_fill_f32<<<cdiv((long)Nn * dout, 256), 256, 0, stream>>>(AGG, 0.f, (long)Nn * dout);
      k_aggpool<<<cdiv((long)NEDGE * dout, 256), 256, 0, stream>>>(H_B, csrc, cdst, cmsk,
                                                                   AGG, dout,
                                                                   (long)NEDGE * dout);
      k_score<<<cdiv(Nn, 256), 256, 0, stream>>>(AGG, H_B, pl[i].Wrel, pl[i].Wroot,
                                                 pl[i].brel, SCORE, Nn, dout);
      int kk = n_cur / 2;
      k_topk<<<NGRAPH, 256, 0, stream>>>(SCORE, INV, n_cur, kk);
      k_poolgather<<<cdiv((long)Nn * dout, 256), 256, 0, stream>>>(H_B, SCORE, INV, H_A,
                                                                   n_cur, kk, dout,
                                                                   (long)Nn * dout);
      int* nsrc = (i == 0) ? SRC_B : SRC_C;
      int* ndst = (i == 0) ? DST_B : DST_C;
      int* nmsk = (i == 0) ? MSK_B : MSK_C;
      k_remap<<<cdiv(NEDGE, 256), 256, 0, stream>>>(csrc, cdst, cmsk, INV, nsrc, ndst,
                                                    nmsk, n_cur, kk, NEDGE);
      csrc = nsrc; cdst = ndst; cmsk = nmsk;
      n_cur = kk; hcur = H_A;
    }
  }

  // ---- final fc: out = silu(graph_out@W + b), [8, 512] ----
  k_fc<<<NGRAPH, 512, 0, stream>>>(GOUT, fcW, fcb, (float*)d_out);
}
